// _LTransformerBlock_35639638622251
// MI455X (gfx1250) — matrix operand order repaired and checked by reading, not compiled, not run on hardware
//
#include <hip/hip_runtime.h>
#include <hip/hip_bf16.h>
#include <math.h>
#include <stdint.h>

// ---------------------------------------------------------------------------
// CDNA5 WMMA types (wave32): bf16 16x16x32 with fp32 accumulation.
// ---------------------------------------------------------------------------
typedef __attribute__((ext_vector_type(16))) __bf16 bf16x16;
typedef __attribute__((ext_vector_type(8)))  __bf16 bf16x8;
typedef __attribute__((ext_vector_type(8)))  float  f32x8;

__device__ __forceinline__ f32x8 wmma_bf16(bf16x16 a, bf16x16 b, f32x8 c) {
    return __builtin_amdgcn_wmma_f32_16x16x32_bf16(
        false, a, false, b, (short)0, c, false, false);
}

__device__ __forceinline__ bf16x16 joinv(bf16x8 lo, bf16x8 hi) {
    return __builtin_shufflevector(lo, hi, 0, 1, 2, 3, 4, 5, 6, 7,
                                   8, 9, 10, 11, 12, 13, 14, 15);
}

// ---------------------------------------------------------------------------
// CDNA5 async global->LDS copy (GLOBAL_LOAD_ASYNC_TO_LDS_B128, ASYNCcnt).
// VDST VGPR carries the LDS byte address (low 32 bits of the generic shared
// pointer per the LDS aperture rule); VADDR is the 64-bit global address.
// ---------------------------------------------------------------------------
__device__ __forceinline__ void async_copy_b128(uint32_t lds_addr,
                                                const void* gaddr) {
    asm volatile("global_load_async_to_lds_b128 %0, %1, off"
                 :: "v"(lds_addr), "v"(gaddr) : "memory");
}
__device__ __forceinline__ void wait_async0() {
    asm volatile("s_wait_asynccnt 0" ::: "memory");
}

// Problem constants
#define BB    4
#define NN    1024
#define DD    1024
#define HH    16
#define HD    64
#define ROWS  (BB * NN)   // 4096
#define DPAD  96          // per-head dim padded 65 -> 96 (multiple of 32)
#define VPAD  80          // attn@V output width padded 65 -> 80
#define CATLD 1056        // concat width padded 1040 -> 1056 (multiple of 32)

// ---------------------------------------------------------------------------
// Block-wide sum reduction (256 threads).
// ---------------------------------------------------------------------------
__device__ __forceinline__ float block_sum(float v, float* red) {
    int tid = threadIdx.x;
    __syncthreads();
    red[tid] = v;
    __syncthreads();
    for (int o = 128; o > 0; o >>= 1) {
        if (tid < o) red[tid] += red[tid + o];
        __syncthreads();
    }
    return red[0];
}

// ---------------------------------------------------------------------------
// Weight demote+transpose: W[K][N] fp32 -> Wt[Np][Kp] bf16 (zero padded).
// ---------------------------------------------------------------------------
__global__ __launch_bounds__(256) void wt_convert_kernel(
    const float* __restrict__ W, __bf16* __restrict__ Wt,
    int K, int N, int Kp, int Np)
{
    int idx = blockIdx.x * 256 + threadIdx.x;
    if (idx >= Np * Kp) return;
    int n = idx / Kp, k = idx % Kp;
    float v = (n < N && k < K) ? W[(size_t)k * N + n] : 0.0f;
    Wt[idx] = (__bf16)v;
}

// ---------------------------------------------------------------------------
// Lorentz LayerNorm + lift -> bf16 (consumed only by GEMM A-operand).
// ---------------------------------------------------------------------------
__global__ __launch_bounds__(256) void ln_lift_kernel(
    const float* __restrict__ in, const float* __restrict__ gamma,
    const float* __restrict__ beta, __bf16* __restrict__ outp)
{
    __shared__ float red[256];
    int row = blockIdx.x, tid = threadIdx.x;
    const float* xr = in + (size_t)row * DD;
    __bf16* orow = outp + (size_t)row * DD;

    float s = 0.f, s2 = 0.f;
    for (int j = 1 + tid; j < DD; j += 256) {
        float v = xr[j];
        s += v; s2 += v * v;
    }
    float mean = block_sum(s, red) * (1.0f / 1023.0f);
    float ex2  = block_sum(s2, red) * (1.0f / 1023.0f);
    float inv  = rsqrtf(ex2 - mean * mean + 1e-5f);

    float ss = 0.f;
    for (int j = 1 + tid; j < DD; j += 256) {
        float v = (xr[j] - mean) * inv * gamma[j - 1] + beta[j - 1];
        orow[j] = (__bf16)v;
        ss += v * v;
    }
    float tot = block_sum(ss, red);
    if (tid == 0) orow[0] = (__bf16)sqrtf(tot + 1.0f);
}

// ---------------------------------------------------------------------------
// GEMM: C[M,N] = A[M,K]@W[K,N] + bias.  A bf16 (K-major), Bt bf16 transposed
// [Np][Kp] (K-major).  K multiple of 32 (zero padded), M multiple of 128.
// Block tile 128x64, 8 waves, wave tile 32x32.  A slab staged into LDS with
// CDNA5 async global->LDS b128 copies (ASYNCcnt); B fragments are direct
// b128 loads from the L2-resident transposed weights.
// ---------------------------------------------------------------------------
__global__ __launch_bounds__(256) void gemm_bf16_kernel(
    const __bf16* __restrict__ A, int lda,
    const __bf16* __restrict__ Bt, int ldb,
    const float* __restrict__ bias,
    float* __restrict__ C, int ldc,
    int N, int K)
{
    __shared__ __bf16 As[128][32];
    int tid  = threadIdx.x;
    int m0   = blockIdx.y * 128;
    int n0   = blockIdx.x * 64;
    int wave = tid >> 5, lane = tid & 31, half = lane >> 4, lr = lane & 15;
    int wm   = (wave >> 1) * 32;
    int wn   = (wave & 1) * 32;

    int sr = tid >> 1;             // staging row 0..127
    int sc = (tid & 1) * 16;       // staging col 0 or 16
    const __bf16* aptr = A + (size_t)(m0 + sr) * lda + sc;
    uint32_t lds0 = (uint32_t)(uintptr_t)&As[sr][sc];
    uint32_t lds1 = lds0 + 16;

    f32x8 acc[2][2] = {};
    int ksteps = K >> 5;

    for (int ks = 0; ks < ksteps; ks++) {
        int k0 = ks << 5;
        // Async-stage this thread's 32B of the A slab into LDS.
        async_copy_b128(lds0, aptr + k0);
        async_copy_b128(lds1, aptr + k0 + 8);
        if (ks + 1 < ksteps) __builtin_prefetch(aptr + k0 + 32, 0, 1);
        wait_async0();
        __syncthreads();

        bf16x16 afr[2], bfr[2];
#pragma unroll
        for (int t = 0; t < 2; t++) {
            const __bf16* ap = &As[wm + t * 16 + lr][half * 8];
            afr[t] = joinv(*(const bf16x8*)ap, *(const bf16x8*)(ap + 16));
        }
#pragma unroll
        for (int u = 0; u < 2; u++) {
            const __bf16* bp =
                Bt + (size_t)(n0 + wn + u * 16 + lr) * ldb + k0 + half * 8;
            bfr[u] = joinv(*(const bf16x8*)bp, *(const bf16x8*)(bp + 16));
        }
#pragma unroll
        for (int t = 0; t < 2; t++)
#pragma unroll
            for (int u = 0; u < 2; u++)
                acc[t][u] = wmma_bf16(afr[t], bfr[u], acc[t][u]);
        __syncthreads();
    }

    // Epilogue: C layout VGPR j -> M = j + 8*half, col = lr.
#pragma unroll
    for (int t = 0; t < 2; t++) {
#pragma unroll
        for (int u = 0; u < 2; u++) {
            int col = n0 + wn + u * 16 + lr;
            if (col >= N) continue;
            float bv = bias[col];
#pragma unroll
            for (int j = 0; j < 8; j++) {
                int row = m0 + wm + t * 16 + j + 8 * half;
                C[(size_t)row * ldc + col] = acc[t][u][j] + bv;
            }
        }
    }
}

// ---------------------------------------------------------------------------
// Per-head lift (q/k): lin[(b*N+n), h*64+d] -> out[((b*H+h)*N+n), 96] bf16.
// q: scale=0.25 (folds 2/sqrt(HD)) and time negated (Lorentz signature).
// ---------------------------------------------------------------------------
__global__ __launch_bounds__(256) void make_heads_kernel(
    const float* __restrict__ lin, __bf16* __restrict__ outp,
    float scale, int negtime)
{
    int g = blockIdx.x * 256 + threadIdx.x;   // (b*H + h)*N + n
    if (g >= BB * HH * NN) return;
    int n = g & (NN - 1);
    int h = (g >> 10) & (HH - 1);
    int b = g >> 14;
    const float* src = lin + ((size_t)(b * NN + n)) * DD + h * HD;
    __bf16* dst = outp + (size_t)g * DPAD;

    float ss = 0.f;
    for (int d = 0; d < HD; d++) { float v = src[d]; ss += v * v; }
    float t = sqrtf(ss + 1.0f);
    dst[0] = (__bf16)(scale * (negtime ? -t : t));
    for (int d = 0; d < HD; d++) dst[1 + d] = (__bf16)(scale * src[d]);
    for (int i = 1 + HD; i < DPAD; i++) dst[i] = (__bf16)0.0f;
}

// ---------------------------------------------------------------------------
// V lift, stored TRANSPOSED: vt[bh][d(80)][n(1024)] bf16, zero padded, so
// attn@V B-fragments are contiguous b128 loads.
// ---------------------------------------------------------------------------
__global__ __launch_bounds__(256) void make_vt_kernel(
    const float* __restrict__ lin, __bf16* __restrict__ vt)
{
    int g = blockIdx.x * 256 + threadIdx.x;
    if (g >= BB * HH * NN) return;
    int n = g & (NN - 1);
    int h = (g >> 10) & (HH - 1);
    int b = g >> 14;
    int bh = b * HH + h;
    const float* src = lin + ((size_t)(b * NN + n)) * DD + h * HD;
    float ss = 0.f;
    for (int d = 0; d < HD; d++) { float v = src[d]; ss += v * v; }
    float t = sqrtf(ss + 1.0f);
    __bf16* col = vt + (size_t)bh * VPAD * NN + n;
    col[0] = (__bf16)t;
    for (int d = 0; d < HD; d++) col[(size_t)(1 + d) * NN] = (__bf16)src[d];
    for (int d = 1 + HD; d < VPAD; d++) col[(size_t)d * NN] = (__bf16)0.0f;
}

// ---------------------------------------------------------------------------
// Attention per (b*H+h, 16-row q block): S = q'k^T + 0.25 (scale folded into
// q), softmax in LDS, mid = attn@V via WMMA, Lorentz normalize, write bf16
// concat row (stride 1056, pads pre-zeroed by memset).
// ---------------------------------------------------------------------------
__global__ __launch_bounds__(256) void attention_kernel(
    const __bf16* __restrict__ qh, const __bf16* __restrict__ kh,
    const __bf16* __restrict__ vt, __bf16* __restrict__ cat)
{
    __shared__ __bf16 S[16][1024];
    __shared__ float  mid[16][VPAD];
    __shared__ float  red2[16][16];
    __shared__ float  rstat[16];

    int tid = threadIdx.x, wave = tid >> 5, lane = tid & 31;
    int half = lane >> 4, lr = lane & 15;
    int bh = blockIdx.x;
    int q0 = blockIdx.y * 16;

    // q fragments for the 3 K-chunks, kept in VGPRs.
    bf16x16 afr[3];
    const __bf16* qbase = qh + ((size_t)bh * NN + q0) * DPAD;
#pragma unroll
    for (int kc = 0; kc < 3; kc++) {
        const __bf16* qp = qbase + (size_t)lr * DPAD + kc * 32 + half * 8;
        afr[kc] = joinv(*(const bf16x8*)qp, *(const bf16x8*)(qp + 16));
    }

    // Scores: 64 key tiles round-robin over 8 waves.
    for (int nt = wave; nt < 64; nt += 8) {
        f32x8 acc = {};
        const __bf16* kp = kh + ((size_t)bh * NN + nt * 16 + lr) * DPAD;
#pragma unroll
        for (int kc = 0; kc < 3; kc++) {
            const __bf16* bp = kp + kc * 32 + half * 8;
            bf16x16 bfr = joinv(*(const bf16x8*)bp, *(const bf16x8*)(bp + 16));
            acc = wmma_bf16(afr[kc], bfr, acc);
        }
#pragma unroll
        for (int j = 0; j < 8; j++)
            S[j + 8 * half][nt * 16 + lr] = (__bf16)(acc[j] + 0.25f);
    }
    __syncthreads();

    // Softmax over 1024 keys, 16 threads per row.
    int r = tid >> 4, c0 = (tid & 15) * 64;
    float mx = -3.0e38f;
    for (int c = c0; c < c0 + 64; c++) mx = fmaxf(mx, (float)S[r][c]);
    red2[r][tid & 15] = mx;
    __syncthreads();
    if (tid < 16) {
        float m = red2[tid][0];
        for (int i = 1; i < 16; i++) m = fmaxf(m, red2[tid][i]);
        rstat[tid] = m;
    }
    __syncthreads();
    float rm = rstat[r];
    float sum = 0.f;
    for (int c = c0; c < c0 + 64; c++) {
        float e = expf((float)S[r][c] - rm);
        S[r][c] = (__bf16)e;
        sum += e;
    }
    red2[r][tid & 15] = sum;
    __syncthreads();
    if (tid < 16) {
        float s = 0.f;
        for (int i = 0; i < 16; i++) s += red2[tid][i];
        rstat[tid] = 1.0f / s;
    }
    __syncthreads();
    float invs = rstat[r];
    for (int c = c0; c < c0 + 64; c++)
        S[r][c] = (__bf16)((float)S[r][c] * invs);
    __syncthreads();

    // mid = attn(16x1024) @ v(1024x80): 5 output tiles on waves 0..4.
    if (wave < 5) {
        f32x8 acc = {};
        int d = wave * 16 + lr;
#pragma unroll 1
        for (int kc = 0; kc < 32; kc++) {
            const __bf16* ap = &S[lr][kc * 32 + half * 8];
            bf16x16 a2 = joinv(*(const bf16x8*)ap, *(const bf16x8*)(ap + 16));
            const __bf16* bp =
                vt + ((size_t)bh * VPAD + d) * NN + kc * 32 + half * 8;
            bf16x16 b2 = joinv(*(const bf16x8*)bp, *(const bf16x8*)(bp + 16));
            acc = wmma_bf16(a2, b2, acc);
        }
#pragma unroll
        for (int j = 0; j < 8; j++)
            mid[j + 8 * half][wave * 16 + lr] = acc[j];
    }
    __syncthreads();

    // Lorentz normalize + scatter into concat (bf16, stride 1056).
    {
        int rr = tid >> 4, t16 = tid & 15;
        float part = 0.f;
        for (int d = t16; d < 65; d += 16) {
            float v = mid[rr][d];
            part += (d == 0) ? -v * v : v * v;
        }
        red2[rr][t16] = part;
        __syncthreads();
        if (tid < 16) {
            float lin = 0.f;
            for (int i = 0; i < 16; i++) lin += red2[tid][i];
            rstat[tid] = 1.0f / sqrtf(fmaxf(-lin, 1e-8f));
        }
        __syncthreads();
        float invd = rstat[rr];
        int b = bh >> 4, h = bh & 15;
        __bf16* crow = cat + ((size_t)(b * NN) + q0 + rr) * CATLD + h * 65;
        for (int d = t16; d < 65; d += 16) crow[d] = (__bf16)(mid[rr][d] * invd);
    }
}

// ---------------------------------------------------------------------------
// Lorentz residual: z = x + w*lift(yspace); out = z / sqrt(max(-<z,z>_L,eps))
// ---------------------------------------------------------------------------
__global__ __launch_bounds__(256) void lresnet_kernel(
    const float* __restrict__ x, const float* __restrict__ yspace,
    const float* __restrict__ wsc, float* __restrict__ outp)
{
    __shared__ float red[256];
    int row = blockIdx.x, tid = threadIdx.x;
    const float* yr = yspace + (size_t)row * (DD - 1);
    const float* xr = x + (size_t)row * DD;
    float* orow = outp + (size_t)row * DD;
    float w = wsc[0];

    float ss = 0.f;
    for (int j = tid; j < DD - 1; j += 256) { float v = yr[j]; ss += v * v; }
    float t = sqrtf(block_sum(ss, red) + 1.0f);

    float lin = 0.f;
    for (int j = tid; j < DD; j += 256) {
        float yv = (j == 0) ? t : yr[j - 1];
        float z = xr[j] + w * yv;
        orow[j] = z;
        lin += (j == 0) ? -z * z : z * z;
    }
    float lint = block_sum(lin, red);
    float invd = 1.0f / sqrtf(fmaxf(-lint, 1e-8f));
    for (int j = tid; j < DD; j += 256) orow[j] *= invd;
}

// ---------------------------------------------------------------------------
// Exact GELU + lift -> bf16 rows of 4096 (time in col 0).
// ---------------------------------------------------------------------------
__global__ __launch_bounds__(256) void gelu_lift_kernel(
    const float* __restrict__ fc, __bf16* __restrict__ hg)
{
    __shared__ float red[256];
    int row = blockIdx.x, tid = threadIdx.x;
    const float* fr = fc + (size_t)row * 4095;
    __bf16* orow = hg + (size_t)row * 4096;
    float ss = 0.f;
    for (int j = tid; j < 4095; j += 256) {
        float v = fr[j];
        float g = 0.5f * v * (1.0f + erff(v * 0.70710678118654752f));
        orow[1 + j] = (__bf16)g;
        ss += g * g;
    }
    float tot = block_sum(ss, red);
    if (tid == 0) orow[0] = (__bf16)sqrtf(tot + 1.0f);
}

// ---------------------------------------------------------------------------
// Host orchestration
// ---------------------------------------------------------------------------
extern "C" void kernel_launch(void* const* d_in, const int* in_sizes, int n_in,
                              void* d_out, int out_size, void* d_ws, size_t ws_size,
                              hipStream_t stream) {
    const float* x   = (const float*)d_in[0];
    const float* g1  = (const float*)d_in[1];
    const float* b1  = (const float*)d_in[2];
    const float* Wq  = (const float*)d_in[3];
    const float* bq  = (const float*)d_in[4];
    const float* Wk  = (const float*)d_in[5];
    const float* bk  = (const float*)d_in[6];
    const float* Wv  = (const float*)d_in[7];
    const float* bv  = (const float*)d_in[8];
    const float* Wo  = (const float*)d_in[9];
    const float* bo  = (const float*)d_in[10];
    const float* g2  = (const float*)d_in[11];
    const float* b2  = (const float*)d_in[12];
    const float* Wfc = (const float*)d_in[13];
    const float* bfc = (const float*)d_in[14];
    const float* Wpj = (const float*)d_in[15];
    const float* bpj = (const float*)d_in[16];
    const float* w1  = (const float*)d_in[17];
    const float* w2  = (const float*)d_in[18];
    float* outp = (float*)d_out;
    char*  ws   = (char*)d_ws;

    // Workspace layout (bytes); fc reuses the dead lxb/qkv/qh region.
    const size_t OFF_LXB  = 0;           // bf16 4096x1024   =  8,388,608
    const size_t OFF_QLIN = 8388608;     // f32  4096x1024   = 16,777,216
    const size_t OFF_KLIN = 25165824;
    const size_t OFF_VLIN = 41943040;
    const size_t OFF_QH   = 58720256;    // bf16 65536x96    = 12,582,912
    const size_t OFF_KH   = 71303168;
    const size_t OFF_VT   = 83886080;    // bf16 64x80x1024  = 10,485,760
    const size_t OFF_CAT  = 94371840;    // bf16 4096x1056   =  8,650,752
    const size_t OFF_AX   = 103022592;   // f32  4096x1023   = 16,760,832
    const size_t OFF_X1   = 119783424;   // f32  4096x1024   = 16,777,216
    const size_t OFF_HLNB = 136560640;   // bf16 4096x1024   =  8,388,608
    const size_t OFF_HGB  = 144949248;   // bf16 4096x4096   = 33,554,432
    const size_t OFF_PJ   = 178503680;   // f32  4096x1023   = 16,760,832
    const size_t OFF_WQT  = 195264512;   // bf16 1024x1024   =  2,097,152
    const size_t OFF_WKT  = 197361664;
    const size_t OFF_WVT  = 199458816;
    const size_t OFF_WOT  = 201555968;   // bf16 1024x1056   =  2,162,688
    const size_t OFF_WFCT = 203718656;   // bf16 4096x1024   =  8,388,608
    const size_t OFF_WPJT = 212107264;   // bf16 1024x4096   =  8,388,608
    const size_t OFF_FC   = 0;           // f32 4096x4095 = 67,092,480 (reuse)
    // total required: 220,495,872 bytes

    __bf16* lxb  = (__bf16*)(ws + OFF_LXB);
    float*  qlin = (float*)(ws + OFF_QLIN);
    float*  klin = (float*)(ws + OFF_KLIN);
    float*  vlin = (float*)(ws + OFF_VLIN);
    __bf16* qhp  = (__bf16*)(ws + OFF_QH);
    __bf16* khp  = (__bf16*)(ws + OFF_KH);
    __bf16* vtp  = (__bf16*)(ws + OFF_VT);
    __bf16* cat  = (__bf16*)(ws + OFF_CAT);
    float*  ax   = (float*)(ws + OFF_AX);
    float*  x1   = (float*)(ws + OFF_X1);
    __bf16* hlnb = (__bf16*)(ws + OFF_HLNB);
    __bf16* hgb  = (__bf16*)(ws + OFF_HGB);
    float*  pj   = (float*)(ws + OFF_PJ);
    __bf16* Wqt  = (__bf16*)(ws + OFF_WQT);
    __bf16* Wkt  = (__bf16*)(ws + OFF_WKT);
    __bf16* Wvt  = (__bf16*)(ws + OFF_WVT);
    __bf16* Wot  = (__bf16*)(ws + OFF_WOT);
    __bf16* Wfct = (__bf16*)(ws + OFF_WFCT);
    __bf16* Wpjt = (__bf16*)(ws + OFF_WPJT);
    float*  fcb  = (float*)(ws + OFF_FC);

    dim3 blk(256);

    // 0) One-time weight demote+transpose into padded [Np][Kp] bf16.
    wt_convert_kernel<<<(1024 * 1024 + 255) / 256, blk, 0, stream>>>(Wq, Wqt, 1024, 1024, 1024, 1024);
    wt_convert_kernel<<<(1024 * 1024 + 255) / 256, blk, 0, stream>>>(Wk, Wkt, 1024, 1024, 1024, 1024);
    wt_convert_kernel<<<(1024 * 1024 + 255) / 256, blk, 0, stream>>>(Wv, Wvt, 1024, 1024, 1024, 1024);
    wt_convert_kernel<<<(1024 * 1056 + 255) / 256, blk, 0, stream>>>(Wo, Wot, 1040, 1023, 1056, 1024);
    wt_convert_kernel<<<(4096 * 1024 + 255) / 256, blk, 0, stream>>>(Wfc, Wfct, 1024, 4095, 1024, 4096);
    wt_convert_kernel<<<(1024 * 4096 + 255) / 256, blk, 0, stream>>>(Wpj, Wpjt, 4096, 1023, 4096, 1024);

    // 1) LN1 + lift -> bf16
    ln_lift_kernel<<<ROWS, blk, 0, stream>>>(x, g1, b1, lxb);

    // 2) QKV projections (M=4096, K=1024, N=1024)
    dim3 gqkv(1024 / 64, ROWS / 128);
    gemm_bf16_kernel<<<gqkv, blk, 0, stream>>>(lxb, 1024, Wqt, 1024, bq, qlin, 1024, 1024, 1024);
    gemm_bf16_kernel<<<gqkv, blk, 0, stream>>>(lxb, 1024, Wkt, 1024, bk, klin, 1024, 1024, 1024);
    gemm_bf16_kernel<<<gqkv, blk, 0, stream>>>(lxb, 1024, Wvt, 1024, bv, vlin, 1024, 1024, 1024);

    // 3) Head lifts: q scaled/negated, k plain, v transposed.
    make_heads_kernel<<<(BB * HH * NN) / 256, blk, 0, stream>>>(qlin, qhp, 0.25f, 1);
    make_heads_kernel<<<(BB * HH * NN) / 256, blk, 0, stream>>>(klin, khp, 1.0f, 0);
    make_vt_kernel<<<(BB * HH * NN) / 256, blk, 0, stream>>>(vlin, vtp);

    // 4) Attention -> bf16 concat (zero pads first; deterministic each call).
    hipMemsetAsync(cat, 0, (size_t)ROWS * CATLD * sizeof(__bf16), stream);
    dim3 gat(BB * HH, NN / 16);
    attention_kernel<<<gat, blk, 0, stream>>>(qhp, khp, vtp, cat);

    // 5) Output projection (K=1056 padded, N=1023)
    dim3 gwo(16, ROWS / 128);
    gemm_bf16_kernel<<<gwo, blk, 0, stream>>>(cat, CATLD, Wot, CATLD, bo, ax, 1023, 1023, CATLD);

    // 6) Residual 1
    lresnet_kernel<<<ROWS, blk, 0, stream>>>(x, ax, w1, x1);

    // 7) LN2 + lift -> bf16
    ln_lift_kernel<<<ROWS, blk, 0, stream>>>(x1, g2, b2, hlnb);

    // 8) FFN up (N=4095)
    dim3 gfc(64, ROWS / 128);
    gemm_bf16_kernel<<<gfc, blk, 0, stream>>>(hlnb, 1024, Wfct, 1024, bfc, fcb, 4095, 4095, 1024);

    // 9) GELU + lift -> bf16 (4096 wide)
    gelu_lift_kernel<<<ROWS, blk, 0, stream>>>(fcb, hgb);

    // 10) FFN down (K=4096, N=1023)
    gemm_bf16_kernel<<<gwo, blk, 0, stream>>>(hgb, 4096, Wpjt, 4096, bpj, pj, 1023, 1023, 4096);

    // 11) Residual 2 -> output
    lresnet_kernel<<<ROWS, blk, 0, stream>>>(x1, pj, w2, outp);
}
